// MambaPyTorch_1176821039844
// MI455X (gfx1250) — matrix operand order repaired and checked by reading, not compiled, not run on hardware
//
#include <hip/hip_runtime.h>
#include <stdint.h>

// ---------------------------------------------------------------------------
// Mamba-like block, fused pipeline for gfx1250 (MI455X), wave32 + WMMA bf16.
//   B=4, T=8192, D=512, H=1024, N=16, K_conv=4
// GEMMs: 32x64 tile per wave, ping-pong fragment buffers (loads rename into
// dead registers -> no v_mov rotation, no WMMA->VALU hazard NOPs).
// ---------------------------------------------------------------------------

#define BSZ   4
#define TLEN  8192
#define DM    512
#define HID_  1024
#define NST   16
#define MTOT  (BSZ * TLEN)          // 32768 rows

typedef __attribute__((ext_vector_type(16))) __bf16 v16bf;
typedef __attribute__((ext_vector_type(8)))  float  v8f;

// ---------------- helpers ----------------

__device__ __forceinline__ uint16_t f2bf(float f) {
  uint32_t u = __float_as_uint(f);
  u += 0x7FFFu + ((u >> 16) & 1u);        // round-to-nearest-even
  return (uint16_t)(u >> 16);
}
__device__ __forceinline__ float bf2f(uint16_t h) {
  return __uint_as_float(((uint32_t)h) << 16);
}

union FragCast {
  struct { uint4 lo, hi; } q;
  v16bf v;
};

// A-matrix fragment (16x32 bf16): lane<16 -> K[0..7]+K[16..23], lane>=16 -> +8.
// Caller passes p already offset by (lane>>4)*8 elements; we read +0 and +16.
__device__ __forceinline__ v16bf load_frag_a(const uint16_t* p) {
  FragCast f;
  f.q.lo = *(const uint4*)(p);
  f.q.hi = *(const uint4*)(p + 16);
  return f.v;
}

// B-matrix fragment (32x16 bf16): lane<16 -> K[0..7]+K[16..23], lane>=16 -> +8.
// Caller passes p already offset by (lane>>4)*8 elements; we read +0 and +16.
__device__ __forceinline__ v16bf load_frag_b(const uint16_t* p) {
  FragCast f;
  f.q.lo = *(const uint4*)(p);
  f.q.hi = *(const uint4*)(p + 16);
  return f.v;
}

__device__ __forceinline__ v8f wmma_bf16(v16bf a, v16bf b, v8f c) {
  return __builtin_amdgcn_wmma_f32_16x16x32_bf16(
      false, a, false, b, (short)0, c, false, false);
}

// One K-step (K=32) worth of fragments for a 32x64 tile.
struct FragSet {
  v16bf a0, a1, b0, b1, b2, b3;
};

__device__ __forceinline__ FragSet load_set(
    const uint16_t* __restrict__ a0p, const uint16_t* __restrict__ a1p,
    const uint16_t* __restrict__ b0p, const uint16_t* __restrict__ b1p,
    const uint16_t* __restrict__ b2p, const uint16_t* __restrict__ b3p,
    int k0) {
  FragSet f;
  f.a0 = load_frag_a(a0p + k0);
  f.a1 = load_frag_a(a1p + k0);
  f.b0 = load_frag_b(b0p + k0);
  f.b1 = load_frag_b(b1p + k0);
  f.b2 = load_frag_b(b2p + k0);
  f.b3 = load_frag_b(b3p + k0);
  return f;
}

__device__ __forceinline__ void wmma_set(const FragSet& f, v8f acc[8]) {
  acc[0] = wmma_bf16(f.a0, f.b0, acc[0]);
  acc[1] = wmma_bf16(f.a0, f.b1, acc[1]);
  acc[2] = wmma_bf16(f.a0, f.b2, acc[2]);
  acc[3] = wmma_bf16(f.a0, f.b3, acc[3]);
  acc[4] = wmma_bf16(f.a1, f.b0, acc[4]);
  acc[5] = wmma_bf16(f.a1, f.b1, acc[5]);
  acc[6] = wmma_bf16(f.a1, f.b2, acc[6]);
  acc[7] = wmma_bf16(f.a1, f.b3, acc[7]);
}

// 32(M) x 64(N) x K tile per wave; ping-pong double buffering (K/32 even).
template <int K>
__device__ __forceinline__ void wmma_tile_32x64(
    const uint16_t* __restrict__ a0p, const uint16_t* __restrict__ a1p,
    const uint16_t* __restrict__ b0p, const uint16_t* __restrict__ b1p,
    const uint16_t* __restrict__ b2p, const uint16_t* __restrict__ b3p,
    v8f acc[8]) {
  static_assert((K / 32) % 2 == 0, "K/32 must be even");
  FragSet f0 = load_set(a0p, a1p, b0p, b1p, b2p, b3p, 0);
  FragSet f1 = load_set(a0p, a1p, b0p, b1p, b2p, b3p, 32);
#pragma unroll 1
  for (int k0 = 0; k0 < K - 64; k0 += 64) {
    wmma_set(f0, acc);
    f0 = load_set(a0p, a1p, b0p, b1p, b2p, b3p, k0 + 64);   // renames, no copies
    wmma_set(f1, acc);
    f1 = load_set(a0p, a1p, b0p, b1p, b2p, b3p, k0 + 96);
  }
  wmma_set(f0, acc);
  wmma_set(f1, acc);
}

// ---------------- kernel 0: fp32 -> bf16 conversion ----------------

__global__ void k_f2bf(const float* __restrict__ in, uint16_t* __restrict__ out, int n) {
  int i = blockIdx.x * blockDim.x + threadIdx.x;
  if (i < n) out[i] = f2bf(in[i]);
}

// ---------------- kernel 1: inp = x @ W_in^T + b_in ; gate/silu epilogue ----
// X: [M,512] bf16, W: [2048,512] bf16. Wave: 32(M) x 64(N) tile.

__global__ void __launch_bounds__(256, 1)
k_gemm_in(const uint16_t* __restrict__ Xbf,
          const uint16_t* __restrict__ Wbf,
          const float* __restrict__ b_in,
          uint16_t* __restrict__ gate,
          uint16_t* __restrict__ upd) {
  const int lane   = threadIdx.x & 31;
  const int wave   = threadIdx.x >> 5;
  const int task   = blockIdx.x * 8 + wave;          // 32768 tasks
  const int mtile  = task >> 5;                      // 0..1023 (32 rows each)
  const int nstrip = task & 31;                      // 0..31   (64 cols each)

  const int nl    = lane & 15;
  const int aoff  = (lane >> 4) * 8;
  const int boff  = (lane >> 4) * 8;
  const int row0  = mtile * 32 + nl;
  const int nbase = nstrip * 64;

  const uint16_t* a0p = Xbf + (size_t)row0 * DM + aoff;
  const uint16_t* a1p = a0p + (size_t)16 * DM;
  const uint16_t* b0p = Wbf + (size_t)(nbase +  0 + nl) * DM + boff;
  const uint16_t* b1p = Wbf + (size_t)(nbase + 16 + nl) * DM + boff;
  const uint16_t* b2p = Wbf + (size_t)(nbase + 32 + nl) * DM + boff;
  const uint16_t* b3p = Wbf + (size_t)(nbase + 48 + nl) * DM + boff;

  v8f acc[8] = {v8f{}, v8f{}, v8f{}, v8f{}, v8f{}, v8f{}, v8f{}, v8f{}};
  wmma_tile_32x64<DM>(a0p, a1p, b0p, b1p, b2p, b3p, acc);

  // Epilogue. Strip (64 cols) lies entirely in gate half (n<1024) or update half.
  const bool isGate = (nbase < HID_);
  uint16_t* dst = isGate ? gate : upd;
  const int mrow = (lane >> 4) * 8;
#pragma unroll
  for (int ms = 0; ms < 2; ++ms) {
#pragma unroll
    for (int t = 0; t < 4; ++t) {
      const int n    = nbase + t * 16 + nl;
      const float bv = b_in[n];
      const int hcol = isGate ? n : (n - HID_);
      const v8f& a = acc[ms * 4 + t];
#pragma unroll
      for (int r = 0; r < 8; ++r) {
        const int m   = mtile * 32 + ms * 16 + mrow + r;
        const float v = a[r] + bv;
        const float sg = 1.0f / (1.0f + __expf(-v));
        dst[(size_t)m * HID_ + hcol] = f2bf(isGate ? sg : v * sg);
      }
    }
  }
}

// ---------------- kernel 2: depthwise conv (k=4) + Bu = u @ Bmat -----------
// One wave per (b,t). Lane handles 32 channels (stride-32 interleave).
// Bu stored (b, n, t)-major so the scan kernel reads contiguously.

__global__ void k_conv_bu(const uint16_t* __restrict__ upd,
                          const float* __restrict__ conv_w,   // [H,1,4]
                          const float* __restrict__ conv_b,   // [H]
                          const float* __restrict__ Bmat,     // [H,16]
                          float* __restrict__ Bu) {           // [B*16, T]
  const int lane = threadIdx.x & 31;
  const int wave = threadIdx.x >> 5;
  const int wg   = blockIdx.x * 8 + wave;           // 0..32767
  const int b    = wg >> 13;
  const int t    = wg & (TLEN - 1);

  const uint16_t* base = upd + (size_t)b * TLEN * HID_;
  float acc[NST];
#pragma unroll
  for (int n = 0; n < NST; ++n) acc[n] = 0.0f;

  for (int c = 0; c < 32; ++c) {
    const int h = lane + 32 * c;
    const float w0 = conv_w[h * 4 + 0], w1 = conv_w[h * 4 + 1];
    const float w2 = conv_w[h * 4 + 2], w3 = conv_w[h * 4 + 3];
    float u = conv_b[h];
    // y[t] = sum_j w[j] * update[t + j - 2], zero-padded
    if (t >= 2)       u += w0 * bf2f(base[(size_t)(t - 2) * HID_ + h]);
    if (t >= 1)       u += w1 * bf2f(base[(size_t)(t - 1) * HID_ + h]);
    u += w2 * bf2f(base[(size_t)t * HID_ + h]);
    if (t + 1 < TLEN) u += w3 * bf2f(base[(size_t)(t + 1) * HID_ + h]);

    const float* brow = Bmat + h * NST;
#pragma unroll
    for (int n = 0; n < NST; ++n) acc[n] += u * brow[n];
  }

  // cross-lane butterfly reduce (all lanes end with full sums)
#pragma unroll
  for (int off = 16; off >= 1; off >>= 1) {
#pragma unroll
    for (int n = 0; n < NST; ++n) acc[n] += __shfl_xor(acc[n], off, 32);
  }
  float outv = 0.0f;
#pragma unroll
  for (int n = 0; n < NST; ++n) outv = (lane == n) ? acc[n] : outv;
  if (lane < NST)
    Bu[((size_t)(b * NST + lane)) * TLEN + t] = outv;
}

// ---------------- kernel 3: inclusive cumsum over T (in place) -------------
// One wave per (b,n) sequence of 8192 contiguous floats.

__global__ void k_scan(float* __restrict__ Bu) {
  const int lane = threadIdx.x;
  float* p = Bu + (size_t)blockIdx.x * TLEN;
  float carry = 0.0f;
  for (int c = 0; c < TLEN / 32; ++c) {
    float v = p[c * 32 + lane];
#pragma unroll
    for (int off = 1; off < 32; off <<= 1) {
      float tmp = __shfl_up(v, off, 32);
      if (lane >= off) v += tmp;
    }
    v += carry;
    p[c * 32 + lane] = v;
    carry = __shfl(v, 31, 32);
  }
}

// ---------------- kernel 4: out_h = gate * (hcum @ A^T) --------------------
// Block per (b,t); hcum row (16 floats) staged in LDS; A is L2-resident.

__global__ void k_outh(const uint16_t* __restrict__ gate,
                       const float* __restrict__ hc,     // [B*16, T]
                       const float* __restrict__ A,      // [H,16]
                       uint16_t* __restrict__ outh) {    // [M,H] bf16
  const int bt = blockIdx.x;                  // == global row m
  const int b  = bt >> 13;
  const int t  = bt & (TLEN - 1);
  __shared__ float sh[NST];
  if (threadIdx.x < NST)
    sh[threadIdx.x] = hc[((size_t)(b * NST + threadIdx.x)) * TLEN + t];
  __syncthreads();

  const size_t m = (size_t)bt;
#pragma unroll
  for (int c = 0; c < 4; ++c) {
    const int h = threadIdx.x + 256 * c;
    const float* ar = A + h * NST;
    float d = 0.0f;
#pragma unroll
    for (int n = 0; n < NST; ++n) d += sh[n] * ar[n];
    const float g = bf2f(gate[m * HID_ + h]);
    outh[m * HID_ + h] = f2bf(g * d);
  }
}

// ---------------- kernel 5: y = out_h @ W_out^T + b_out --------------------
// M=32768, N=512, K=1024. Wave: 32(M) x 64(N) tile; fp32 output.

__global__ void __launch_bounds__(256, 1)
k_gemm_out(const uint16_t* __restrict__ Xbf,   // [M,1024]
           const uint16_t* __restrict__ Wbf,   // [512,1024]
           const float* __restrict__ b_out,
           float* __restrict__ y) {            // [M,512]
  const int lane   = threadIdx.x & 31;
  const int wave   = threadIdx.x >> 5;
  const int task   = blockIdx.x * 8 + wave;          // 8192 tasks
  const int mtile  = task >> 3;                      // 0..1023 (32 rows each)
  const int nstrip = task & 7;                       // 0..7    (64 cols each)

  const int nl    = lane & 15;
  const int aoff  = (lane >> 4) * 8;
  const int boff  = (lane >> 4) * 8;
  const int row0  = mtile * 32 + nl;
  const int nbase = nstrip * 64;

  const uint16_t* a0p = Xbf + (size_t)row0 * HID_ + aoff;
  const uint16_t* a1p = a0p + (size_t)16 * HID_;
  const uint16_t* b0p = Wbf + (size_t)(nbase +  0 + nl) * HID_ + boff;
  const uint16_t* b1p = Wbf + (size_t)(nbase + 16 + nl) * HID_ + boff;
  const uint16_t* b2p = Wbf + (size_t)(nbase + 32 + nl) * HID_ + boff;
  const uint16_t* b3p = Wbf + (size_t)(nbase + 48 + nl) * HID_ + boff;

  v8f acc[8] = {v8f{}, v8f{}, v8f{}, v8f{}, v8f{}, v8f{}, v8f{}, v8f{}};
  wmma_tile_32x64<HID_>(a0p, a1p, b0p, b1p, b2p, b3p, acc);

  const int mrow = (lane >> 4) * 8;
#pragma unroll
  for (int ms = 0; ms < 2; ++ms) {
#pragma unroll
    for (int t = 0; t < 4; ++t) {
      const int n    = nbase + t * 16 + nl;
      const float bv = b_out[n];
      const v8f& a = acc[ms * 4 + t];
#pragma unroll
      for (int r = 0; r < 8; ++r) {
        const int m = mtile * 32 + ms * 16 + mrow + r;
        y[(size_t)m * DM + n] = a[r] + bv;
      }
    }
  }
}

// ---------------- host-side launch -----------------------------------------

extern "C" void kernel_launch(void* const* d_in, const int* in_sizes, int n_in,
                              void* d_out, int out_size, void* d_ws, size_t ws_size,
                              hipStream_t stream) {
  const float* x      = (const float*)d_in[0];   // (B,T,512)
  const float* W_in   = (const float*)d_in[1];   // (2048,512)
  const float* b_in   = (const float*)d_in[2];   // (2048)
  const float* conv_w = (const float*)d_in[3];   // (1024,1,4)
  const float* conv_b = (const float*)d_in[4];   // (1024)
  const float* A      = (const float*)d_in[5];   // (1024,16)
  const float* Bmat   = (const float*)d_in[6];   // (1024,16)
  const float* W_out  = (const float*)d_in[7];   // (512,1024)
  const float* b_out  = (const float*)d_in[8];   // (512)
  float* y = (float*)d_out;

  // workspace layout (bytes)
  char* w = (char*)d_ws;
  const size_t n_x    = (size_t)MTOT * DM;          // 16,777,216
  const size_t n_win  = (size_t)2 * HID_ * DM;      //  1,048,576
  const size_t n_wout = (size_t)DM * HID_;          //    524,288
  const size_t n_gh   = (size_t)MTOT * HID_;        // 33,554,432

  uint16_t* x_bf    = (uint16_t*)w;                 w += n_x * 2;
  uint16_t* win_bf  = (uint16_t*)w;                 w += n_win * 2;
  uint16_t* wout_bf = (uint16_t*)w;                 w += n_wout * 2;
  uint16_t* gate_bf = (uint16_t*)w;                 w += n_gh * 2;
  uint16_t* upd_bf  = (uint16_t*)w;                 w += n_gh * 2;
  uint16_t* outh_bf = (uint16_t*)w;                 w += n_gh * 2;
  float*    Bu      = (float*)w;                    w += (size_t)BSZ * NST * TLEN * 4;

  // 0) fp32 -> bf16 conversions
  k_f2bf<<<(int)((n_x    + 255) / 256), 256, 0, stream>>>(x,     x_bf,    (int)n_x);
  k_f2bf<<<(int)((n_win  + 255) / 256), 256, 0, stream>>>(W_in,  win_bf,  (int)n_win);
  k_f2bf<<<(int)((n_wout + 255) / 256), 256, 0, stream>>>(W_out, wout_bf, (int)n_wout);

  // 1) input GEMM + gate/silu epilogue: 1024 mtiles x 32 nstrips, 8 waves/block
  k_gemm_in<<<4096, 256, 0, stream>>>(x_bf, win_bf, b_in, gate_bf, upd_bf);

  // 2) depthwise conv + Bu projection: one wave per (b,t)
  k_conv_bu<<<4096, 256, 0, stream>>>(upd_bf, conv_w, conv_b, Bmat, Bu);

  // 3) cumulative sum over T: 64 contiguous sequences
  k_scan<<<64, 32, 0, stream>>>(Bu);

  // 4) out_h = gate * (hcum @ A^T)
  k_outh<<<MTOT, 256, 0, stream>>>(gate_bf, Bu, A, outh_bf);

  // 5) output GEMM: 1024 mtiles x 8 nstrips, 8 waves/block
  k_gemm_out<<<1024, 256, 0, stream>>>(outh_bf, wout_bf, b_out, y);
}
